// SelfAttention_43860206026896
// MI455X (gfx1250) — matrix operand order repaired and checked by reading, not compiled, not run on hardware
//
#include <hip/hip_runtime.h>

// Problem constants
#define L_DIM  4096
#define C_DIM  1024
#define CQ_DIM 128

typedef __attribute__((ext_vector_type(16))) _Float16 v16h;
typedef __attribute__((ext_vector_type(8)))  _Float16 v8h;
typedef __attribute__((ext_vector_type(8)))  float    v8f;

// ---------------------------------------------------------------------------
// WMMA fragment loads, per CDNA5 ISA 7.12.2 (16-bit A 16x32, B 32x16, wave32)
// ---------------------------------------------------------------------------
__device__ __forceinline__ v16h load_frag_a(const _Float16* __restrict__ A, int lda,
                                            int m0, int k0, int lane) {
  const int m = m0 + (lane & 15);
  const int khalf = (lane < 16) ? 0 : 16;
  const _Float16* p = A + (size_t)m * lda + (k0 + khalf);
  return *(const v16h*)(p);
}

__device__ __forceinline__ v16h load_frag_b(const _Float16* __restrict__ Bt, int ldb,
                                            int n0, int k0, int lane) {
  const int n = n0 + (lane & 15);
  const int kbase = (lane < 16) ? 0 : 16;
  return *(const v16h*)(Bt + (size_t)n * ldb + (k0 + kbase));
}

// ---------------------------------------------------------------------------
// Register-blocked mainloop: one wave computes a 32x64 block (2 M-tiles x
// 4 N-tiles, 8 f32 accumulators). Inner loop keeps only ONE B fragment live
// at a time (spill avoidance): per K-step of 32, 2 A-frags are reused across
// 4 B-frags -> 8 WMMAs from 12 b128 loads. Each B frag feeds two WMMAs that
// write different accumulators (no D->A/B hazard).
// ---------------------------------------------------------------------------
#define BM_TILES 2
#define BN_TILES 4

__device__ __forceinline__ void wmma_block_nt(const _Float16* __restrict__ A, int lda,
                                              const _Float16* __restrict__ Bt, int ldb,
                                              int m0, int n0, int K, int lane,
                                              v8f acc[BM_TILES][BN_TILES]) {
#pragma unroll
  for (int i = 0; i < BM_TILES; ++i)
#pragma unroll
    for (int j = 0; j < BN_TILES; ++j)
      acc[i][j] = (v8f){};

  for (int k0 = 0; k0 < K; k0 += 32) {
    v16h a0 = load_frag_a(A, lda, m0, k0, lane);
    v16h a1 = load_frag_a(A, lda, m0 + 16, k0, lane);
#pragma unroll
    for (int j = 0; j < BN_TILES; ++j) {
      v16h b = load_frag_b(Bt, ldb, n0 + 16 * j, k0, lane);
      acc[0][j] = __builtin_amdgcn_wmma_f32_16x16x32_f16(
          false, a0, false, b, (short)0, acc[0][j], false, false);
      acc[1][j] = __builtin_amdgcn_wmma_f32_16x16x32_f16(
          false, a1, false, b, (short)0, acc[1][j], false, false);
    }
  }
}

// C/D tile layout: lane l -> N = l%16, rows M = (l<16?0:8)+r for r=0..7 = c[r]
// ---------------------------------------------------------------------------

__global__ void cvt_f32_to_f16_kernel(const float* __restrict__ src,
                                      _Float16* __restrict__ dst, int n) {
  int i = blockIdx.x * blockDim.x + threadIdx.x;
  if (i < n) dst[i] = (_Float16)src[i];
}

// Projection GEMM: D[m,n] = sum_k A[m,k]*Bt[n,k] + bias, f16 output.
// bias_on_m=0 -> bias[n] (Q/K proj), bias_on_m=1 -> bias[m] (V proj).
__global__ __launch_bounds__(256, 1)
void gemm_proj_kernel(const _Float16* __restrict__ A, int lda,
                      const _Float16* __restrict__ Bt, int ldb,
                      const float* __restrict__ bias, int bias_on_m,
                      _Float16* __restrict__ D, int ldd,
                      int nblk_n, int K) {
  const int wave = blockIdx.x * (blockDim.x >> 5) + (threadIdx.x >> 5);
  const int lane = threadIdx.x & 31;
  const int m0 = (wave / nblk_n) * (16 * BM_TILES);
  const int n0 = (wave % nblk_n) * (16 * BN_TILES);

  v8f acc[BM_TILES][BN_TILES];
  wmma_block_nt(A, lda, Bt, ldb, m0, n0, K, lane, acc);

  const int nl = lane & 15;
  const int mh = (lane < 16) ? 0 : 8;
#pragma unroll
  for (int i = 0; i < BM_TILES; ++i) {
    const int mb = m0 + 16 * i + mh;
#pragma unroll
    for (int j = 0; j < BN_TILES; ++j) {
      const int n = n0 + 16 * j + nl;
      const float bn = bias_on_m ? 0.0f : bias[n];
#pragma unroll
      for (int r = 0; r < 8; ++r) {
        float v = acc[i][j][r] + (bias_on_m ? bias[mb + r] : bn);
        D[(size_t)(mb + r) * ldd + n] = (_Float16)v;
      }
    }
  }
}

// q2[j] = sum_m Qt[j,m]^2
__global__ void sumsq_kernel(const _Float16* __restrict__ Qt, float* __restrict__ q2) {
  int j = blockIdx.x * blockDim.x + threadIdx.x;
  const _Float16* row = Qt + (size_t)j * CQ_DIM;
  float s = 0.0f;
#pragma unroll 8
  for (int m = 0; m < CQ_DIM; ++m) {
    float v = (float)row[m];
    s += v * v;
  }
  q2[j] = s;
}

// energy[j,i] = exp(-relu(g) * max(q2[j] + k2[i] - 2*(Qt@Kt^T)[j,i], 0))
__global__ __launch_bounds__(256, 1)
void energy_kernel(const _Float16* __restrict__ Qt,
                   const _Float16* __restrict__ Kt,
                   const float* __restrict__ q2,
                   const float* __restrict__ k2,
                   const float* __restrict__ gamma_att,
                   float* __restrict__ att) {
  const int wave = blockIdx.x * (blockDim.x >> 5) + (threadIdx.x >> 5);
  const int lane = threadIdx.x & 31;
  const int nblk_n = L_DIM / (16 * BN_TILES);
  const int m0 = (wave / nblk_n) * (16 * BM_TILES);
  const int n0 = (wave % nblk_n) * (16 * BN_TILES);

  v8f acc[BM_TILES][BN_TILES];
  wmma_block_nt(Qt, CQ_DIM, Kt, CQ_DIM, m0, n0, CQ_DIM, lane, acc);

  const float g = fmaxf(gamma_att[0], 0.0f);
  const int nl = lane & 15;
  const int mh = (lane < 16) ? 0 : 8;
#pragma unroll
  for (int i = 0; i < BM_TILES; ++i) {
    const int mb = m0 + 16 * i + mh;
#pragma unroll
    for (int j = 0; j < BN_TILES; ++j) {
      const int n = n0 + 16 * j + nl;
      const float k2n = k2[n];
#pragma unroll
      for (int r = 0; r < 8; ++r) {
        float sq = fmaxf(q2[mb + r] + k2n - 2.0f * acc[i][j][r], 0.0f);
        att[(size_t)(mb + r) * L_DIM + n] = __expf(-g * sq);
      }
    }
  }
}

// Row softmax over energy (in place, fp32) + f16 copy for the next GEMM.
__global__ void softmax_kernel(float* __restrict__ att, _Float16* __restrict__ attH) {
  const int row = blockIdx.x;
  __shared__ float red[256];
  float* arow = att + (size_t)row * L_DIM;
  float s = 0.0f;
  for (int i = threadIdx.x; i < L_DIM; i += 256) s += __expf(arow[i]);
  red[threadIdx.x] = s;
  __syncthreads();
  for (int off = 128; off > 0; off >>= 1) {
    if (threadIdx.x < off) red[threadIdx.x] += red[threadIdx.x + off];
    __syncthreads();
  }
  const float inv = 1.0f / red[0];
  _Float16* hrow = attH + (size_t)row * L_DIM;
  for (int i = threadIdx.x; i < L_DIM; i += 256) {
    float a = __expf(arow[i]) * inv;
    arow[i] = a;
    hrow[i] = (_Float16)a;
  }
}

// outT[j,c] = gamma * sum_i att[j,i] * V[c,i] + x[j,c]
__global__ __launch_bounds__(256, 1)
void out_gemm_kernel(const _Float16* __restrict__ attH,
                     const _Float16* __restrict__ Vh,
                     const float* __restrict__ x,
                     const float* __restrict__ gamma,
                     float* __restrict__ out) {
  const int wave = blockIdx.x * (blockDim.x >> 5) + (threadIdx.x >> 5);
  const int lane = threadIdx.x & 31;
  const int nblk_n = C_DIM / (16 * BN_TILES);
  const int m0 = (wave / nblk_n) * (16 * BM_TILES);
  const int n0 = (wave % nblk_n) * (16 * BN_TILES);

  v8f acc[BM_TILES][BN_TILES];
  wmma_block_nt(attH, L_DIM, Vh, L_DIM, m0, n0, L_DIM, lane, acc);

  const float g = gamma[0];
  const int nl = lane & 15;
  const int mh = (lane < 16) ? 0 : 8;
#pragma unroll
  for (int i = 0; i < BM_TILES; ++i) {
    const int mb = m0 + 16 * i + mh;
#pragma unroll
    for (int j = 0; j < BN_TILES; ++j) {
      const int n = n0 + 16 * j + nl;
#pragma unroll
      for (int r = 0; r < 8; ++r) {
        size_t idx = (size_t)(mb + r) * C_DIM + n;
        out[idx] = g * acc[i][j][r] + x[idx];
      }
    }
  }
}

__global__ void tail_kernel(const float* __restrict__ gamma,
                            const float* __restrict__ gamma_att,
                            float* __restrict__ out_tail) {
  if (threadIdx.x == 0) {
    out_tail[0] = gamma[0];
    out_tail[1] = gamma_att[0];
  }
}

// ---------------------------------------------------------------------------

extern "C" void kernel_launch(void* const* d_in, const int* in_sizes, int n_in,
                              void* d_out, int out_size, void* d_ws, size_t ws_size,
                              hipStream_t stream) {
  (void)in_sizes; (void)n_in; (void)out_size; (void)ws_size;

  const float* x         = (const float*)d_in[0]; // [L, C]
  const float* Wq        = (const float*)d_in[1]; // [CQ, C]
  const float* bq        = (const float*)d_in[2]; // [CQ]
  const float* Wk        = (const float*)d_in[3]; // [CQ, C]
  const float* bk        = (const float*)d_in[4]; // [CQ]
  const float* Wv        = (const float*)d_in[5]; // [C, C]
  const float* bv        = (const float*)d_in[6]; // [C]
  const float* gamma     = (const float*)d_in[7]; // [1]
  const float* gamma_att = (const float*)d_in[8]; // [1]

  float* out_main = (float*)d_out;                              // [L, C]
  float* att      = out_main + (size_t)L_DIM * C_DIM;           // [L, L]
  float* out_tail = att + (size_t)L_DIM * L_DIM;                // gamma, gamma_att

  // Workspace carve-up (f16 operand copies + transposed activations)
  char* w = (char*)d_ws;
  _Float16* xh   = (_Float16*)w; w += (size_t)L_DIM * C_DIM * 2;   // [L, C]
  _Float16* Wqh  = (_Float16*)w; w += (size_t)CQ_DIM * C_DIM * 2;  // [CQ, C]
  _Float16* Wkh  = (_Float16*)w; w += (size_t)CQ_DIM * C_DIM * 2;  // [CQ, C]
  _Float16* Wvh  = (_Float16*)w; w += (size_t)C_DIM * C_DIM * 2;   // [C, C]
  _Float16* Qt   = (_Float16*)w; w += (size_t)L_DIM * CQ_DIM * 2;  // [L, CQ]
  _Float16* Kt   = (_Float16*)w; w += (size_t)L_DIM * CQ_DIM * 2;  // [L, CQ]
  _Float16* Vh   = (_Float16*)w; w += (size_t)C_DIM * L_DIM * 2;   // [C, L]
  _Float16* attH = (_Float16*)w; w += (size_t)L_DIM * L_DIM * 2;   // [L, L]
  float*    q2   = (float*)w;    w += (size_t)L_DIM * 4;
  float*    k2   = (float*)w;    w += (size_t)L_DIM * 4;

  const int TPB = 256;                 // 8 waves per block
  const int WPB = TPB / 32;

  // 1) f32 -> f16 operand conversion
  cvt_f32_to_f16_kernel<<<(L_DIM * C_DIM) / TPB, TPB, 0, stream>>>(x, xh, L_DIM * C_DIM);
  cvt_f32_to_f16_kernel<<<(CQ_DIM * C_DIM) / TPB, TPB, 0, stream>>>(Wq, Wqh, CQ_DIM * C_DIM);
  cvt_f32_to_f16_kernel<<<(CQ_DIM * C_DIM) / TPB, TPB, 0, stream>>>(Wk, Wkh, CQ_DIM * C_DIM);
  cvt_f32_to_f16_kernel<<<(C_DIM * C_DIM) / TPB, TPB, 0, stream>>>(Wv, Wvh, C_DIM * C_DIM);

  // 2) Projections (register-blocked WMMA GEMMs, 32x64 block per wave)
  // Qt[j,m] = sum_c x[j,c]*Wq[m,c] + bq[m]   : M=L, N=CQ, K=C
  {
    int nblk_n = CQ_DIM / (16 * BN_TILES);           // 2
    int waves  = (L_DIM / (16 * BM_TILES)) * nblk_n; // 256
    gemm_proj_kernel<<<waves / WPB, TPB, 0, stream>>>(xh, C_DIM, Wqh, C_DIM, bq, 0,
                                                      Qt, CQ_DIM, nblk_n, C_DIM);
    gemm_proj_kernel<<<waves / WPB, TPB, 0, stream>>>(xh, C_DIM, Wkh, C_DIM, bk, 0,
                                                      Kt, CQ_DIM, nblk_n, C_DIM);
  }
  // Vh[c,i] = sum_cc Wv[c,cc]*x[i,cc] + bv[c] : M=C, N=L, K=C
  {
    int nblk_n = L_DIM / (16 * BN_TILES);            // 64
    int waves  = (C_DIM / (16 * BM_TILES)) * nblk_n; // 2048
    gemm_proj_kernel<<<waves / WPB, TPB, 0, stream>>>(Wvh, C_DIM, xh, C_DIM, bv, 1,
                                                      Vh, L_DIM, nblk_n, C_DIM);
  }

  // 3) Row squared norms
  sumsq_kernel<<<L_DIM / TPB, TPB, 0, stream>>>(Qt, q2);
  sumsq_kernel<<<L_DIM / TPB, TPB, 0, stream>>>(Kt, k2);

  // 4) Gaussian-kernel energy (WMMA QK^T fused with exp epilogue)
  {
    int waves = (L_DIM / (16 * BM_TILES)) * (L_DIM / (16 * BN_TILES)); // 8192
    energy_kernel<<<waves / WPB, TPB, 0, stream>>>(Qt, Kt, q2, k2, gamma_att, att);
  }

  // 5) Row softmax (in-place fp32 attention + f16 copy)
  softmax_kernel<<<L_DIM, TPB, 0, stream>>>(att, attH);

  // 6) out^T = gamma * (att @ V^T) + x  (WMMA with residual epilogue)
  {
    int waves = (L_DIM / (16 * BM_TILES)) * (C_DIM / (16 * BN_TILES)); // 2048
    out_gemm_kernel<<<waves / WPB, TPB, 0, stream>>>(attH, Vh, x, gamma, out_main);
  }

  // 7) Scalar outputs
  tail_kernel<<<1, 32, 0, stream>>>(gamma, gamma_att, out_tail);
}
